// MultiHeadAttention_49005576847632
// MI455X (gfx1250) — matrix operand order repaired and checked by reading, not compiled, not run on hardware
//
#include <hip/hip_runtime.h>

// ---------------- problem constants ----------------
#define B_SZ   4
#define T_SZ   2048
#define D_SZ   1024
#define H_SZ   16
#define DH_SZ  64
#define BT_SZ  (B_SZ * T_SZ)          // 8192
// softmax works in "raw score" domain; 1/sqrt(64) folded into the exp2 base:
// exp(0.125*x) = 2^(x * 0.125*log2(e))
#define SCALE_LOG2E 0.180336880111120419f
#define LOG2E       1.44269504088896340736f

// ---------------- CDNA5 WMMA types ----------------
typedef __attribute__((ext_vector_type(16))) __bf16 v16bf;
typedef __attribute__((ext_vector_type(8)))  __bf16 v8bf;
typedef __attribute__((ext_vector_type(8)))  float  v8f;

static __device__ __forceinline__ unsigned short f32_to_bf16(float f) {
    union { float f; unsigned int u; } c; c.f = f;
    unsigned int u = c.u;
    u += 0x7FFFu + ((u >> 16) & 1u);        // round-to-nearest-even
    return (unsigned short)(u >> 16);
}

static __device__ __forceinline__ v16bf cat8(v8bf lo, v8bf hi) {
    return __builtin_shufflevector(lo, hi, 0,1,2,3,4,5,6,7,8,9,10,11,12,13,14,15);
}

static __device__ __forceinline__ v8f wmma_bf16(v16bf a, v16bf b, v8f c) {
    // D(f32 16x16) = A(bf16 16x32) * B(bf16 32x16) + C
    return __builtin_amdgcn_wmma_f32_16x16x32_bf16(
        /*neg_a=*/false, a, /*neg_b=*/false, b,
        /*c_mod=*/(short)0, c, /*reuse_a=*/false, /*reuse_b=*/false);
}

// ---------------- f32 -> bf16 cast ----------------
__global__ void cast_f32_bf16(const float* __restrict__ in,
                              unsigned short* __restrict__ out, int n) {
    int i = blockIdx.x * blockDim.x + threadIdx.x;
    int stride = gridDim.x * blockDim.x;
    for (; i < n; i += stride) out[i] = f32_to_bf16(in[i]);
}

// ---------------- GEMM: Y[M,N] = A[M,K] @ W[N,K]^T + bias ----------------
// M = 8192, N = K = 1024 (fixed). One wave computes a 32(M) x 64(N) tile
// (two A fragments share each B fragment -> 8 WMMAs per 12 b128 loads).
// Block = 4 waves (128 threads) -> 128x64 tile per block.
enum { OUT_F32 = 0, OUT_QK = 1, OUT_VT = 2 };

template <int MODE>
__global__ __launch_bounds__(128)
void gemm_bias_bf16(const unsigned short* __restrict__ A,
                    const unsigned short* __restrict__ W,
                    const float* __restrict__ bias,
                    void* __restrict__ outp)
{
    const int lane = threadIdx.x & 31;
    const int wave = threadIdx.x >> 5;
    const int half = lane >> 4;            // 0..1 (wave32 half)
    const int nl   = lane & 15;
    const int m0   = blockIdx.y * 128 + wave * 32;
    const int n0   = blockIdx.x * 64;

    const unsigned short* arow0 = A + (size_t)(m0 + nl) * D_SZ;
    const unsigned short* arow1 = A + (size_t)(m0 + 16 + nl) * D_SZ;
    v8f acc[2][4] = {};

    for (int k0 = 0; k0 < D_SZ; k0 += 32) {
        __builtin_prefetch(arow0 + k0 + 256, 0, 0);              // global_prefetch_b8
        // A fragments (16x32 bf16): lo half K = half*8.., hi half K = 16+half*8..
        v16bf a0 = cat8(*(const v8bf*)(arow0 + k0 + half * 8),
                        *(const v8bf*)(arow0 + k0 + 16 + half * 8));
        v16bf a1 = cat8(*(const v8bf*)(arow1 + k0 + half * 8),
                        *(const v8bf*)(arow1 + k0 + 16 + half * 8));
#pragma unroll
        for (int nt = 0; nt < 4; ++nt) {
            // B fragment (32x16 bf16): lane col n, lo half K = half*8.., hi half K = 16+half*8..
            const unsigned short* brow =
                W + (size_t)(n0 + nt * 16 + nl) * D_SZ + k0 + half * 8;
            v16bf b = cat8(*(const v8bf*)brow, *(const v8bf*)(brow + 16));
            acc[0][nt] = wmma_bf16(a0, b, acc[0][nt]);
            acc[1][nt] = wmma_bf16(a1, b, acc[1][nt]);
        }
    }

    // Epilogue: C layout row m = m0 + rb*16 + r + half*8, col n = n0 + nt*16 + nl
#pragma unroll
    for (int rb = 0; rb < 2; ++rb) {
#pragma unroll
        for (int nt = 0; nt < 4; ++nt) {
            const int n = n0 + nt * 16 + nl;
            const float bv = bias[n];
#pragma unroll
            for (int r = 0; r < 8; ++r) {
                const int m = m0 + rb * 16 + r + half * 8;
                const float v = acc[rb][nt][r] + bv;
                if (MODE == OUT_F32) {
                    ((float*)outp)[(size_t)m * D_SZ + n] = v;
                } else {
                    const int b = m >> 11, t = m & (T_SZ - 1);
                    const int h = n >> 6,  d = n & (DH_SZ - 1);
                    unsigned short* o = (unsigned short*)outp;
                    if (MODE == OUT_QK)      // (B,H,T,Dh)
                        o[(((size_t)(b * H_SZ + h)) * T_SZ + t) * DH_SZ + d] = f32_to_bf16(v);
                    else                     // (B,H,Dh,T)  V transposed
                        o[(((size_t)(b * H_SZ + h)) * DH_SZ + d) * T_SZ + t] = f32_to_bf16(v);
                }
            }
        }
    }
}

// ---------------- causal flash attention ----------------
// One wave per 16 query rows; key blocks of 32; online softmax in f32.
// Exactly one key block straddles the diagonal -> peeled; the main loop is
// mask-free (no v_cmp/v_cndmask in steady state).
__global__ __launch_bounds__(128)
void attention_kernel(const unsigned short* __restrict__ Q,   // (B,H,T,Dh) bf16
                      const unsigned short* __restrict__ K,   // (B,H,T,Dh) bf16
                      const unsigned short* __restrict__ Vt,  // (B,H,Dh,T) bf16
                      unsigned short* __restrict__ attn)      // (B,T,D) bf16
{
    __shared__ unsigned short plds[4][16][32];   // per-wave P transpose tile

    const int lane = threadIdx.x & 31;
    const int wave = threadIdx.x >> 5;
    const int half = lane >> 4;
    const int nl   = lane & 15;
    const int q0   = blockIdx.x * 64 + wave * 16;
    const int h    = blockIdx.y;
    const int b    = blockIdx.z;
    const size_t bh = (size_t)(b * H_SZ + h);

    const unsigned short* Qb = Q  + bh * T_SZ * DH_SZ;
    const unsigned short* Kb = K  + bh * T_SZ * DH_SZ;
    const unsigned short* Vb = Vt + bh * DH_SZ * T_SZ;

    // Q A-fragments for d0 = 0 and 32, loaded once
    const unsigned short* qrow = Qb + (size_t)(q0 + nl) * DH_SZ;
    v16bf qa[2];
#pragma unroll
    for (int dblk = 0; dblk < 2; ++dblk) {
        const int d0 = dblk * 32;
        qa[dblk] = cat8(*(const v8bf*)(qrow + d0 + half * 8),
                        *(const v8bf*)(qrow + d0 + 16 + half * 8));
    }

    v8f o[4] = {};
    float mrow[8], lrow[8];
#pragma unroll
    for (int r = 0; r < 8; ++r) { mrow[r] = -__builtin_inff(); lrow[r] = 0.0f; }

    auto process_block = [&](int k0, bool need_mask) {
        // ---- S[16q x 32k] = Q @ K^T (reduce over Dh=64 -> 2 WMMAs per 16k tile)
        v8f s[2] = {};
#pragma unroll
        for (int kt = 0; kt < 2; ++kt) {
            const int key = k0 + kt * 16 + nl;
#pragma unroll
            for (int dblk = 0; dblk < 2; ++dblk) {
                const unsigned short* krow =
                    Kb + (size_t)key * DH_SZ + dblk * 32 + half * 8;
                v16bf kb = cat8(*(const v8bf*)krow, *(const v8bf*)(krow + 16));
                s[kt] = wmma_bf16(qa[dblk], kb, s[kt]);
            }
        }

        // ---- causal mask (only in the single peeled diagonal block)
        if (need_mask) {
#pragma unroll
            for (int kt = 0; kt < 2; ++kt) {
                const int key = k0 + kt * 16 + nl;
#pragma unroll
                for (int r = 0; r < 8; ++r)
                    if (key > (q0 + r + half * 8)) s[kt][r] = -__builtin_inff();
            }
        }

        // ---- row max over the 16-lane column group (butterfly, stays in group)
        float tmax[8];
#pragma unroll
        for (int r = 0; r < 8; ++r) tmax[r] = fmaxf(s[0][r], s[1][r]);
#pragma unroll
        for (int off = 1; off < 16; off <<= 1)
#pragma unroll
            for (int r = 0; r < 8; ++r)
                tmax[r] = fmaxf(tmax[r], __shfl_xor(tmax[r], off, 32));

        // ---- online softmax update (raw-score domain, scale folded into exp2)
        float psum[8];
#pragma unroll
        for (int r = 0; r < 8; ++r) {
            const float mn = fmaxf(mrow[r], tmax[r]);
            const float sc = exp2f((mrow[r] - mn) * SCALE_LOG2E);
            mrow[r] = mn;
            lrow[r] *= sc;
#pragma unroll
            for (int dt = 0; dt < 4; ++dt) o[dt][r] *= sc;
            const float p0 = exp2f((s[0][r] - mn) * SCALE_LOG2E);
            const float p1 = exp2f((s[1][r] - mn) * SCALE_LOG2E);
            s[0][r] = p0; s[1][r] = p1;
            psum[r] = p0 + p1;
        }
#pragma unroll
        for (int off = 1; off < 16; off <<= 1)
#pragma unroll
            for (int r = 0; r < 8; ++r)
                psum[r] += __shfl_xor(psum[r], off, 32);
#pragma unroll
        for (int r = 0; r < 8; ++r) lrow[r] += psum[r];

        // ---- transpose P (C-layout) -> A-layout through per-wave LDS tile
#pragma unroll
        for (int kt = 0; kt < 2; ++kt)
#pragma unroll
            for (int r = 0; r < 8; ++r)
                plds[wave][r + half * 8][kt * 16 + nl] = f32_to_bf16(s[kt][r]);
        asm volatile("s_wait_dscnt 0" ::: "memory");

        v16bf pa = cat8(*(const v8bf*)&plds[wave][nl][half * 8],
                        *(const v8bf*)&plds[wave][nl][16 + half * 8]);

        // ---- O[16q x 64d] += P @ V (V transposed -> contiguous B-frag loads)
#pragma unroll
        for (int dt = 0; dt < 4; ++dt) {
            const unsigned short* vrow =
                Vb + (size_t)(dt * 16 + nl) * T_SZ + k0 + half * 8;
            v16bf vb = cat8(*(const v8bf*)vrow, *(const v8bf*)(vrow + 16));
            o[dt] = wmma_bf16(pa, vb, o[dt]);
        }
    };

    // full (mask-free) blocks: k0 + 31 <= q0
    const int kfull = ((q0 + 1) >> 5) << 5;
    for (int k0 = 0; k0 < kfull; k0 += 32) process_block(k0, false);
    // single diagonal block with elementwise causal mask
    process_block(kfull, true);

    // ---- normalize and store (B,T,D) bf16
#pragma unroll
    for (int r = 0; r < 8; ++r) {
        const int t = q0 + r + half * 8;
        const float inv = 1.0f / lrow[r];
#pragma unroll
        for (int dt = 0; dt < 4; ++dt) {
            const int d = dt * 16 + nl;
            attn[((size_t)b * T_SZ + t) * D_SZ + h * DH_SZ + d] =
                f32_to_bf16(o[dt][r] * inv);
        }
    }
}

// ---------------- host side ----------------
extern "C" void kernel_launch(void* const* d_in, const int* in_sizes, int n_in,
                              void* d_out, int out_size, void* d_ws, size_t ws_size,
                              hipStream_t stream) {
    (void)in_sizes; (void)n_in; (void)out_size; (void)ws_size;
    const float* x  = (const float*)d_in[0];
    const float* wq = (const float*)d_in[1];
    const float* bq = (const float*)d_in[2];
    const float* wk = (const float*)d_in[3];
    const float* bk = (const float*)d_in[4];
    const float* wv = (const float*)d_in[5];
    const float* bv = (const float*)d_in[6];
    const float* wo = (const float*)d_in[7];
    const float* bo = (const float*)d_in[8];

    char* ws = (char*)d_ws;
    size_t off = 0;
    auto alloc = [&](size_t bytes) {
        void* p = ws + off; off += (bytes + 255) & ~(size_t)255; return p;
    };
    unsigned short* xbf    = (unsigned short*)alloc((size_t)BT_SZ * D_SZ * 2);
    unsigned short* wqbf   = (unsigned short*)alloc((size_t)D_SZ * D_SZ * 2);
    unsigned short* wkbf   = (unsigned short*)alloc((size_t)D_SZ * D_SZ * 2);
    unsigned short* wvbf   = (unsigned short*)alloc((size_t)D_SZ * D_SZ * 2);
    unsigned short* wobf   = (unsigned short*)alloc((size_t)D_SZ * D_SZ * 2);
    unsigned short* qbf    = (unsigned short*)alloc((size_t)BT_SZ * D_SZ * 2);
    unsigned short* kbf    = (unsigned short*)alloc((size_t)BT_SZ * D_SZ * 2);
    unsigned short* vtbf   = (unsigned short*)alloc((size_t)BT_SZ * D_SZ * 2);
    unsigned short* attnbf = (unsigned short*)alloc((size_t)BT_SZ * D_SZ * 2);

    cast_f32_bf16<<<4096, 256, 0, stream>>>(x,  xbf,  BT_SZ * D_SZ);
    cast_f32_bf16<<<1024, 256, 0, stream>>>(wq, wqbf, D_SZ * D_SZ);
    cast_f32_bf16<<<1024, 256, 0, stream>>>(wk, wkbf, D_SZ * D_SZ);
    cast_f32_bf16<<<1024, 256, 0, stream>>>(wv, wvbf, D_SZ * D_SZ);
    cast_f32_bf16<<<1024, 256, 0, stream>>>(wo, wobf, D_SZ * D_SZ);

    dim3 gblk(128);
    dim3 ggrid(D_SZ / 64, BT_SZ / 128);   // (16, 64)
    gemm_bias_bf16<OUT_QK><<<ggrid, gblk, 0, stream>>>(xbf, wqbf, bq, qbf);
    gemm_bias_bf16<OUT_QK><<<ggrid, gblk, 0, stream>>>(xbf, wkbf, bk, kbf);
    gemm_bias_bf16<OUT_VT><<<ggrid, gblk, 0, stream>>>(xbf, wvbf, bv, vtbf);

    attention_kernel<<<dim3(T_SZ / 64, H_SZ, B_SZ), 128, 0, stream>>>(qbf, kbf, vtbf, attnbf);

    gemm_bias_bf16<OUT_F32><<<ggrid, gblk, 0, stream>>>(attnbf, wobf, bo, (float*)d_out);
}
